// LinearAttention_6571299963560
// MI455X (gfx1250) — matrix operand order repaired and checked by reading, not compiled, not run on hardware
//
#include <hip/hip_runtime.h>
#include <hip/hip_bf16.h>
#include <math.h>

// ---------------------------------------------------------------------------
// Types for CDNA5 WMMA (wave32): v_wmma_f32_16x16x32_bf16
// ---------------------------------------------------------------------------
typedef __bf16 bf16;
typedef __attribute__((ext_vector_type(16))) __bf16 bf16x16;
typedef __attribute__((ext_vector_type(8)))  __bf16 bf16x8;
typedef __attribute__((ext_vector_type(8)))  float  f32x8;

__device__ __forceinline__ f32x8 wmma_bf16(bf16x16 a, bf16x16 b, f32x8 c) {
    // (neg_a, A, neg_b, B, c_mod, C, reuse_a, reuse_b)
    return __builtin_amdgcn_wmma_f32_16x16x32_bf16(false, a, false, b,
                                                   (short)0, c, false, false);
}

// A fragment: 16x32 bf16, A row-major at p (lda halfwords).
// Lane L: row m = L&15 ; K range: [kb..kb+15], kb = (L&16)?16:0
__device__ __forceinline__ bf16x16 load_a(const bf16* __restrict__ p, int lda, int lane) {
    int m  = lane & 15;
    int kb = (lane & 16) ? 16 : 0;
    const bf16* q = p + (size_t)m * lda + kb;
    union { bf16x16 v; bf16x8 h[2]; } u;
    u.h[0] = *(const bf16x8*)(q);
    u.h[1] = *(const bf16x8*)(q + 8);
    return u.v;
}

// B fragment: 32x16 bf16, column-contiguous storage: column n lives at p + n*ldb,
// 16 contiguous K halfwords per lane.  Lane L: col n = L&15, K base = (L&16)?16:0.
__device__ __forceinline__ bf16x16 load_b(const bf16* __restrict__ p, int ldb, int lane) {
    int n  = lane & 15;
    int kb = (lane & 16) ? 16 : 0;
    return *(const bf16x16*)(p + (size_t)n * ldb + kb);
}

// ---------------------------------------------------------------------------
// Elementwise fp32 -> bf16
// ---------------------------------------------------------------------------
__global__ void k_f32_to_bf16(const float* __restrict__ src, bf16* __restrict__ dst, int n) {
    int i = blockIdx.x * blockDim.x + threadIdx.x;
    if (i < n) dst[i] = (bf16)src[i];
}

// ---------------------------------------------------------------------------
// Depthwise 3x3 (pad 1) + pack fmap and conv result token-major bf16.
// fmap: [8][256][64][64] f32 ;  Xt/Kdwt: [8][4096][256] bf16
// ---------------------------------------------------------------------------
__global__ void k_dwconv_pack(const float* __restrict__ fmap, const float* __restrict__ wdw,
                              bf16* __restrict__ Xt, bf16* __restrict__ Kdwt) {
    int i = blockIdx.x * blockDim.x + threadIdx.x;   // over 8*256*4096
    if (i >= 8 * 256 * 4096) return;
    int xy = i & 4095;
    int c  = (i >> 12) & 255;
    int b  = i >> 20;
    const float* src = fmap + ((size_t)b * 256 + c) * 4096;
    const float* w   = wdw + c * 9;
    int y = xy >> 6, x = xy & 63;
    float acc = 0.f;
    #pragma unroll
    for (int dy = -1; dy <= 1; ++dy) {
        int yy = y + dy;
        if (yy < 0 || yy > 63) continue;
        #pragma unroll
        for (int dx = -1; dx <= 1; ++dx) {
            int xx = x + dx;
            if (xx < 0 || xx > 63) continue;
            acc += w[(dy + 1) * 3 + (dx + 1)] * src[yy * 64 + xx];
        }
    }
    size_t o = ((size_t)b * 4096 + xy) * 256 + c;
    Xt[o]   = (bf16)src[xy];
    Kdwt[o] = (bf16)acc;
}

// ---------------------------------------------------------------------------
// Generic bf16 WMMA GEMM:  C[M][N] (f32, row-major) = A[M][K] (row-major bf16)
//                          x B (column-contiguous bf16: col n at Bc + n*K)
//
// 128 threads = 4 waves.
//  SPLITK == false : wave tile 64(M) x 64(N); block tile 256 x 64.
//                    16 WMMAs per 16 b128 loads per K-step.
//  SPLITK == true  : requires M == N == 64; all 4 waves compute the same
//                    64x64 tile over disjoint K/4 chunks, partials reduced
//                    through LDS (48 KB).  Used for the context GEMM
//                    (M=N=64, K=4096) to quadruple wave parallelism.
// Optional bias per output row.
// ---------------------------------------------------------------------------
template <bool SPLITK>
__global__ void k_gemm_bf16(const bf16* __restrict__ A, const bf16* __restrict__ Bc,
                            float* __restrict__ C, const float* __restrict__ bias,
                            int M, int N, int K,
                            size_t strideA, size_t strideB, size_t strideC) {
    int batch = blockIdx.z;
    A  += (size_t)batch * strideA;
    Bc += (size_t)batch * strideB;
    C  += (size_t)batch * strideC;

    int lane = threadIdx.x & 31;
    int wave = threadIdx.x >> 5;

    int m0, n0, kbeg, kend;
    if constexpr (SPLITK) {
        m0 = 0; n0 = 0;
        int chunk = K >> 2;
        kbeg = wave * chunk;
        kend = kbeg + chunk;
    } else {
        m0 = blockIdx.x * 256 + wave * 64;
        n0 = blockIdx.y * 64;
        kbeg = 0; kend = K;
        if (m0 >= M) return;
    }

    f32x8 acc[4][4] = {};

    for (int k0 = kbeg; k0 < kend; k0 += 32) {
        bf16x16 a[4];
        #pragma unroll
        for (int i = 0; i < 4; ++i)
            a[i] = load_a(A + (size_t)(m0 + 16 * i) * K + k0, K, lane);
        if (k0 + 32 < kend) {   // prefetch next K tile (global_prefetch_b8)
            __builtin_prefetch(A  + (size_t)m0 * K + k0 + 32, 0, 1);
            __builtin_prefetch(Bc + (size_t)n0 * K + k0 + 32, 0, 1);
        }
        #pragma unroll
        for (int j = 0; j < 4; ++j) {
            bf16x16 b = load_b(Bc + (size_t)(n0 + 16 * j) * K + k0, K, lane);
            #pragma unroll
            for (int i = 0; i < 4; ++i)
                acc[i][j] = wmma_bf16(a[i], b, acc[i][j]);
        }
    }

    int n  = lane & 15;
    int mo = (lane & 16) ? 8 : 0;

    if constexpr (SPLITK) {
        // Reduce the 4 waves' partial 64x64 tiles through LDS.
        __shared__ float red[3][64 * 64];
        if (wave > 0) {
            #pragma unroll
            for (int i = 0; i < 4; ++i)
                #pragma unroll
                for (int j = 0; j < 4; ++j)
                    #pragma unroll
                    for (int r = 0; r < 8; ++r) {
                        int row = 16 * i + mo + r;
                        int col = 16 * j + n;
                        red[wave - 1][row * 64 + col] = acc[i][j][r];
                    }
        }
        __syncthreads();
        if (wave != 0) return;
        #pragma unroll
        for (int i = 0; i < 4; ++i)
            #pragma unroll
            for (int j = 0; j < 4; ++j)
                #pragma unroll
                for (int r = 0; r < 8; ++r) {
                    int row = 16 * i + mo + r;
                    int col = 16 * j + n;
                    int idx = row * 64 + col;
                    acc[i][j][r] += red[0][idx] + red[1][idx] + red[2][idx];
                }
    }

    #pragma unroll
    for (int i = 0; i < 4; ++i) {
        #pragma unroll
        for (int j = 0; j < 4; ++j) {
            #pragma unroll
            for (int r = 0; r < 8; ++r) {
                int row = m0 + 16 * i + mo + r;
                int col = n0 + 16 * j + n;
                float v = acc[i][j][r];
                if (bias) v += bias[row];
                C[(size_t)row * N + col] = v;
            }
        }
    }
}

// ---------------------------------------------------------------------------
// softmax over feature dim (64 per head) per token; *SCALE; out token-major bf16.
// Q: [8][512][4096] f32 ;  QsT: [64][4096][64] bf16
// ---------------------------------------------------------------------------
__global__ void k_softmax_q(const float* __restrict__ Q, bf16* __restrict__ QsT) {
    int idx = blockIdx.x * blockDim.x + threadIdx.x;   // over 64*4096
    if (idx >= 64 * 4096) return;
    int bh = idx >> 12, n = idx & 4095;
    int b = bh >> 3, h = bh & 7;
    const float* src = Q + ((size_t)b * 512 + h * 64) * 4096 + n;
    float v[64];
    float mx = -3.0e38f;
    #pragma unroll
    for (int d = 0; d < 64; ++d) { v[d] = src[(size_t)d * 4096]; mx = fmaxf(mx, v[d]); }
    float s = 0.f;
    #pragma unroll
    for (int d = 0; d < 64; ++d) { v[d] = expf(v[d] - mx); s += v[d]; }
    float inv = 0.125f / s;                     // SCALE = dh^-0.5 = 1/8
    bf16* dst = QsT + (size_t)idx * 64;
    #pragma unroll
    for (int g = 0; g < 8; ++g) {
        bf16x8 o;
        #pragma unroll
        for (int r = 0; r < 8; ++r) o[r] = (bf16)(v[g * 8 + r] * inv);
        *(bf16x8*)(dst + g * 8) = o;
    }
}

// ---------------------------------------------------------------------------
// softmax over tokens (4096) per k-channel; out channel-major bf16.
// KV: [8][1024][4096] f32 (k = channels 0..511) ; Ks: [8][512][4096] bf16
// ---------------------------------------------------------------------------
__global__ void k_softmax_k(const float* __restrict__ KV, bf16* __restrict__ Ks) {
    int row = blockIdx.x;                     // 0 .. 8*512-1
    int b = row >> 9, r = row & 511;
    const float* src = KV + ((size_t)b * 1024 + r) * 4096;
    bf16* dst = Ks + (size_t)row * 4096;
    __shared__ float red[256];
    int t = threadIdx.x;
    float v[16];
    float mx = -3.0e38f;
    #pragma unroll
    for (int i = 0; i < 16; ++i) { v[i] = src[t + i * 256]; mx = fmaxf(mx, v[i]); }
    red[t] = mx; __syncthreads();
    for (int s = 128; s > 0; s >>= 1) {
        if (t < s) red[t] = fmaxf(red[t], red[t + s]);
        __syncthreads();
    }
    mx = red[0]; __syncthreads();
    float sum = 0.f;
    #pragma unroll
    for (int i = 0; i < 16; ++i) { v[i] = expf(v[i] - mx); sum += v[i]; }
    red[t] = sum; __syncthreads();
    for (int s = 128; s > 0; s >>= 1) {
        if (t < s) red[t] += red[t + s];
        __syncthreads();
    }
    float inv = 1.f / red[0];
    #pragma unroll
    for (int i = 0; i < 16; ++i) dst[t + i * 256] = (bf16)(v[i] * inv);
}

// v channels (512..1023 of KV) fp32 -> bf16, channel-major.
__global__ void k_convert_v(const float* __restrict__ KV, bf16* __restrict__ V) {
    size_t i = (size_t)blockIdx.x * blockDim.x + threadIdx.x;   // over 8*512*4096
    if (i >= (size_t)8 * 512 * 4096) return;
    int n = (int)(i & 4095);
    size_t rr = i >> 12;
    int b = (int)(rr >> 9), r = (int)(rr & 511);
    V[i] = (bf16)KV[((size_t)b * 1024 + 512 + r) * 4096 + n];
}

// context [bh][d][e] f32 -> transposed bf16 [bh][e][d]
__global__ void k_ctx_transpose(const float* __restrict__ Ctx, bf16* __restrict__ CtxT) {
    int i = blockIdx.x * blockDim.x + threadIdx.x;   // over 64*64*64, i = bh*4096 + e*64 + d
    if (i >= 64 * 64 * 64) return;
    int d = i & 63, e = (i >> 6) & 63, bh = i >> 12;
    CtxT[i] = (bf16)Ctx[(size_t)bh * 4096 + d * 64 + e];
}

// ---------------------------------------------------------------------------
// out_head[e][n] = sum_d CtxT[e][d] * QsT[n][d], fused exact GELU, store
// token-major bf16 into Gt[b][n][h*64+e].  M=64, N=4096, K=64 per (bh).
// 128 threads = 4 waves; wave tile 16(e) x 64(n).
// ---------------------------------------------------------------------------
__global__ void k_attn_out_gelu(const bf16* __restrict__ CtxT, const bf16* __restrict__ QsT,
                                bf16* __restrict__ Gt) {
    int bh = blockIdx.x;
    int b = bh >> 3, h = bh & 7;
    int n0 = blockIdx.y * 64;
    int lane = threadIdx.x & 31;
    int wave = threadIdx.x >> 5;
    int e0 = wave * 16;

    const bf16* A  = CtxT + (size_t)bh * 4096 + (size_t)e0 * 64;          // [16 x 64], lda=64
    const bf16* Bc = QsT + (size_t)bh * 4096 * 64 + (size_t)n0 * 64;      // col n at +n*64

    f32x8 acc[4] = {};
    #pragma unroll
    for (int k0 = 0; k0 < 64; k0 += 32) {
        bf16x16 a = load_a(A + k0, 64, lane);
        #pragma unroll
        for (int j = 0; j < 4; ++j) {
            bf16x16 bm = load_b(Bc + (size_t)(16 * j) * 64 + k0, 64, lane);
            acc[j] = wmma_bf16(a, bm, acc[j]);
        }
    }

    int n  = lane & 15;
    int eo = (lane & 16) ? 8 : 0;
    #pragma unroll
    for (int j = 0; j < 4; ++j) {
        int tok = n0 + 16 * j + n;
        bf16* dst = Gt + ((size_t)b * 4096 + tok) * 512 + h * 64 + e0 + eo;
        bf16x8 o;
        #pragma unroll
        for (int r = 0; r < 8; ++r) {
            float x = acc[j][r];
            float g = 0.5f * x * (1.f + erff(x * 0.70710678118654752f));   // exact GELU
            o[r] = (bf16)g;
        }
        *(bf16x8*)dst = o;
    }
}

// ---------------------------------------------------------------------------
// Host side: workspace layout + launch sequence
// ---------------------------------------------------------------------------
static inline size_t align256(size_t x) { return (x + 255) & ~(size_t)255; }

extern "C" void kernel_launch(void* const* d_in, const int* in_sizes, int n_in,
                              void* d_out, int out_size, void* d_ws, size_t ws_size,
                              hipStream_t stream) {
    (void)in_sizes; (void)n_in; (void)out_size; (void)ws_size;
    const float* fmap = (const float*)d_in[0];   // [8][256][64][64]
    const float* wq   = (const float*)d_in[1];   // [512][256]
    const float* wdw  = (const float*)d_in[2];   // [256][1][3][3]
    const float* wpw  = (const float*)d_in[3];   // [1024][256]
    const float* wout = (const float*)d_in[4];   // [256][512]
    const float* bout = (const float*)d_in[5];   // [256]
    float* out = (float*)d_out;                  // [8][256][64][64]

    char* ws = (char*)d_ws;
    size_t off = 0;
    auto take = [&](size_t bytes) { char* p = ws + off; off += align256(bytes); return p; };

    bf16*  Xt     = (bf16*)take((size_t)8 * 4096 * 256 * 2);    // token-major fmap
    bf16*  Kdwt   = (bf16*)take((size_t)8 * 4096 * 256 * 2);    // token-major dwconv
    bf16*  wq_b   = (bf16*)take((size_t)512 * 256 * 2);
    bf16*  wpw_b  = (bf16*)take((size_t)1024 * 256 * 2);
    bf16*  wout_b = (bf16*)take((size_t)256 * 512 * 2);
    float* Q      = (float*)take((size_t)8 * 512 * 4096 * 4);   // channel-major q
    float* KV     = (float*)take((size_t)8 * 1024 * 4096 * 4);  // channel-major kv
    bf16*  QsT    = (bf16*)take((size_t)64 * 4096 * 64 * 2);    // token-major softmaxed q
    bf16*  Ks     = (bf16*)take((size_t)8 * 512 * 4096 * 2);    // channel-major softmaxed k
    bf16*  Vb     = (bf16*)take((size_t)8 * 512 * 4096 * 2);    // channel-major v (bf16)
    float* Ctx    = (float*)take((size_t)64 * 64 * 64 * 4);
    bf16*  CtxT   = (bf16*)take((size_t)64 * 64 * 64 * 2);
    bf16*  Gt     = (bf16*)Q;   // alias: Q fully consumed before Gt is written

    // 1) weights -> bf16
    k_f32_to_bf16<<<(512 * 256 + 255) / 256, 256, 0, stream>>>(wq, wq_b, 512 * 256);
    k_f32_to_bf16<<<(1024 * 256 + 255) / 256, 256, 0, stream>>>(wpw, wpw_b, 1024 * 256);
    k_f32_to_bf16<<<(256 * 512 + 255) / 256, 256, 0, stream>>>(wout, wout_b, 256 * 512);

    // 2) depthwise conv + token-major pack
    k_dwconv_pack<<<(8 * 256 * 4096 + 255) / 256, 256, 0, stream>>>(fmap, wdw, Xt, Kdwt);

    // 3) Q = wq @ X        (M=512, N=4096, K=256, per batch)
    k_gemm_bf16<false><<<dim3(2, 64, 8), 128, 0, stream>>>(
        wq_b, Xt, Q, nullptr, 512, 4096, 256,
        0, (size_t)4096 * 256, (size_t)512 * 4096);

    // 4) KV = wpw @ Kdw    (M=1024, N=4096, K=256, per batch)
    k_gemm_bf16<false><<<dim3(4, 64, 8), 128, 0, stream>>>(
        wpw_b, Kdwt, KV, nullptr, 1024, 4096, 256,
        0, (size_t)4096 * 256, (size_t)1024 * 4096);

    // 5) softmax over feature dim, *scale, token-major bf16
    k_softmax_q<<<(64 * 4096 + 255) / 256, 256, 0, stream>>>(Q, QsT);

    // 6) softmax over token dim for k
    k_softmax_k<<<8 * 512, 256, 0, stream>>>(KV, Ks);

    // 7) v -> bf16
    k_convert_v<<<(8 * 512 * 4096 + 255) / 256, 256, 0, stream>>>(KV, Vb);

    // 8) context = k_head @ v_headT   (M=64, N=64, K=4096, per bh=64; split-K x4)
    k_gemm_bf16<true><<<dim3(1, 1, 64), 128, 0, stream>>>(
        Ks, Vb, Ctx, nullptr, 64, 64, 4096,
        (size_t)64 * 4096, (size_t)64 * 4096, (size_t)64 * 64);

    // 9) ctx transpose + bf16
    k_ctx_transpose<<<(64 * 64 * 64 + 255) / 256, 256, 0, stream>>>(Ctx, CtxT);

    // 10) out = ctxT @ qsT with fused exact GELU -> token-major Gt
    k_attn_out_gelu<<<dim3(64, 64), 128, 0, stream>>>(CtxT, QsT, Gt);

    // 11) final: out = wout @ gelu(out) + bout   (M=256, N=4096, K=512, per batch)
    k_gemm_bf16<false><<<dim3(1, 64, 8), 128, 0, stream>>>(
        wout_b, Gt, out, bout, 256, 4096, 512,
        0, (size_t)4096 * 512, (size_t)256 * 4096);
}
